// GATLayer_60911226191992
// MI455X (gfx1250) — matrix operand order repaired and checked by reading, not compiled, not run on hardware
//
#include <hip/hip_runtime.h>
#include <hip/hip_bf16.h>
#include <cstdint>

typedef __attribute__((ext_vector_type(16))) _Float16 v16h;
typedef __attribute__((ext_vector_type(8)))  _Float16 v8h;
typedef __attribute__((ext_vector_type(8)))  float    v8f;

#define NODES 4096
#define IND   256
#define OD    64
#define NH    4
#define HD    256   /* NH*OD */
#define NEG_SLOPE 0.2f

__device__ __forceinline__ float lrelu(float e) { return e >= 0.0f ? e : NEG_SLOPE * e; }

// ---------------------------------------------------------------------------
// Kernel 1: h = x @ W  via WMMA f16 (16x16 tile per wave, K-loop of 32).
// Writes hT32[col][node] (f32, for exact alpha dots) and hT16[col][node]
// (f16, B-operand layout for the aggregation GEMM: contiguous along node/j).
// ---------------------------------------------------------------------------
__global__ void __launch_bounds__(128) k_gemm_xw(const float* __restrict__ x,
                                                 const float* __restrict__ W,
                                                 _Float16* __restrict__ hT16,
                                                 float* __restrict__ hT32) {
    const int wave = threadIdx.x >> 5;
    const int lane = threadIdx.x & 31;
    const int tile = blockIdx.x * 4 + wave;      // 4096 tiles total
    const int mt   = tile >> 4;                  // 0..255  (node block)
    const int nt   = tile & 15;                  // 0..15   (col block)
    const int m0 = mt * 16, n0 = nt * 16;
    const int lr  = lane & 15;
    const int hi  = lane >> 4;
    const int sel = hi * 8;

    v8f acc = {};
    const float* xr  = x + (size_t)(m0 + lr) * IND;
    const int    col = n0 + lr;

    for (int k0 = 0; k0 < IND; k0 += 32) {
        // A fragment: lane<16 holds K = {0..7,16..23}, lane>=16 holds {8..15,24..31}
        const float4* pa0 = (const float4*)(xr + k0 + sel);
        const float4* pa1 = (const float4*)(xr + k0 + 16 + sel);
        float4 a0 = pa0[0], a1 = pa0[1], a2 = pa1[0], a3 = pa1[1];
        v16h va;
        va[0]=(_Float16)a0.x; va[1]=(_Float16)a0.y; va[2]=(_Float16)a0.z; va[3]=(_Float16)a0.w;
        va[4]=(_Float16)a1.x; va[5]=(_Float16)a1.y; va[6]=(_Float16)a1.z; va[7]=(_Float16)a1.w;
        va[8]=(_Float16)a2.x; va[9]=(_Float16)a2.y; va[10]=(_Float16)a2.z; va[11]=(_Float16)a2.w;
        va[12]=(_Float16)a3.x; va[13]=(_Float16)a3.y; va[14]=(_Float16)a3.z; va[15]=(_Float16)a3.w;

        // B fragment: lane = column, K = {0..7,16..23} (lo) / {8..15,24..31} (hi)
        v16h vb;
        #pragma unroll
        for (int e = 0; e < 16; ++e)
            vb[e] = (_Float16)W[(size_t)(k0 + sel + (e & 7) + (e >> 3) * 16) * HD + col];

        acc = __builtin_amdgcn_wmma_f32_16x16x32_f16(false, va, false, vb,
                                                     (short)0, acc, false, false);
    }

    // C/D layout: lane = column N=lr; VGPR v holds row M = v + hi*8.
    const size_t base = (size_t)col * NODES + (size_t)(m0 + hi * 8);
    float4 lo = make_float4(acc[0], acc[1], acc[2], acc[3]);
    float4 hi4 = make_float4(acc[4], acc[5], acc[6], acc[7]);
    *(float4*)(hT32 + base)     = lo;
    *(float4*)(hT32 + base + 4) = hi4;
    v8h h8;
    #pragma unroll
    for (int v = 0; v < 8; ++v) h8[v] = (_Float16)acc[v];
    *(v8h*)(hT16 + base) = h8;
}

// ---------------------------------------------------------------------------
// Kernel 2: alpha_src[h][c], alpha_dst[h][c] = <h[c, h,:], a_src/dst[h,:]>
// ---------------------------------------------------------------------------
__global__ void k_alpha(const float* __restrict__ hT32,
                        const float* __restrict__ a_src,
                        const float* __restrict__ a_dst,
                        float* __restrict__ asrc, float* __restrict__ adst) {
    const int c = blockIdx.x * blockDim.x + threadIdx.x;
    #pragma unroll
    for (int h = 0; h < NH; ++h) {
        float s1 = 0.0f, s2 = 0.0f;
        for (int d = 0; d < OD; ++d) {
            float v = hT32[(size_t)(h * OD + d) * NODES + c];
            s1 += v * a_src[h * OD + d];
            s2 += v * a_dst[h * OD + d];
        }
        asrc[h * NODES + c] = s1;
        adst[h * NODES + c] = s2;
    }
}

// ---------------------------------------------------------------------------
// Kernel 3: per-row masked-softmax stats. LeakyReLU is monotonic, so
// m = lrelu(asrc_i + max_{j: adj=1} adst_j). One adj pass via LDS cache.
// (Also warms adj into the 192MB L2 for the aggregation pass.)
// ---------------------------------------------------------------------------
__global__ void __launch_bounds__(256) k_stats(const int* __restrict__ adj,
                                               const float* __restrict__ asrc,
                                               const float* __restrict__ adst,
                                               float* __restrict__ mrow,
                                               float* __restrict__ rinv) {
    __shared__ int   sadj[NODES];
    __shared__ float red[NH * 256];
    __shared__ float bm[NH];
    const int i = blockIdx.x, tid = threadIdx.x;
    const int* arow = adj + (size_t)i * NODES;

    float mx[NH];
    #pragma unroll
    for (int h = 0; h < NH; ++h) mx[h] = -__builtin_inff();

    for (int j = tid; j < NODES; j += 256) {
        int a = arow[j];
        sadj[j] = a;
        if (a) {
            #pragma unroll
            for (int h = 0; h < NH; ++h)
                mx[h] = fmaxf(mx[h], adst[h * NODES + j]);
        }
    }
    #pragma unroll
    for (int h = 0; h < NH; ++h) red[h * 256 + tid] = mx[h];
    __syncthreads();
    for (int s = 128; s > 0; s >>= 1) {
        if (tid < s) {
            #pragma unroll
            for (int h = 0; h < NH; ++h)
                red[h * 256 + tid] = fmaxf(red[h * 256 + tid], red[h * 256 + tid + s]);
        }
        __syncthreads();
    }
    if (tid < NH) {
        float md = red[tid * 256];
        bm[tid] = (md > -__builtin_inff()) ? lrelu(asrc[tid * NODES + i] + md) : 0.0f;
    }
    __syncthreads();

    float as[NH], bmh[NH], sm[NH];
    #pragma unroll
    for (int h = 0; h < NH; ++h) {
        as[h] = asrc[h * NODES + i];
        bmh[h] = bm[h];
        sm[h] = 0.0f;
    }
    for (int j = tid; j < NODES; j += 256) {
        if (sadj[j]) {
            #pragma unroll
            for (int h = 0; h < NH; ++h)
                sm[h] += __expf(lrelu(as[h] + adst[h * NODES + j]) - bmh[h]);
        }
    }
    #pragma unroll
    for (int h = 0; h < NH; ++h) red[h * 256 + tid] = sm[h];
    __syncthreads();
    for (int s = 128; s > 0; s >>= 1) {
        if (tid < s) {
            #pragma unroll
            for (int h = 0; h < NH; ++h)
                red[h * 256 + tid] += red[h * 256 + tid + s];
        }
        __syncthreads();
    }
    if (tid < NH) {
        float s = red[tid * 256];
        mrow[tid * NODES + i] = bm[tid];
        rinv[tid * NODES + i] = (s > 0.0f) ? 1.0f / s : 0.0f;
    }
}

// ---------------------------------------------------------------------------
// Kernel 4: out[h, i0:i0+32, :] = softmax-weighted aggregation via WMMA.
// Block = 4 waves = 4 heads sharing adj tiles; each wave handles TWO 16-row
// i-tiles so each B fragment (hT16) feeds 2 WMMAs -> halved L2 traffic.
// P-tiles are built directly in the WMMA A-fragment lane layout.
// ---------------------------------------------------------------------------
__global__ void __launch_bounds__(128) k_aggregate(const int* __restrict__ adj,
                                                   const _Float16* __restrict__ hT16,
                                                   const float* __restrict__ asrc,
                                                   const float* __restrict__ adst,
                                                   const float* __restrict__ mrow,
                                                   const float* __restrict__ rinv,
                                                   float* __restrict__ out) {
    const int h    = threadIdx.x >> 5;     // wave id = head
    const int lane = threadIdx.x & 31;
    const int lr   = lane & 15;
    const int hi   = lane >> 4;
    const int sel  = hi * 8;
    const int i0   = blockIdx.x * 32;      // 32 rows per block
    const int iA   = i0 + lr;
    const int iB   = i0 + 16 + lr;

    const float asA = asrc[h * NODES + iA];
    const float mA  = mrow[h * NODES + iA];
    const float asB = asrc[h * NODES + iB];
    const float mB  = mrow[h * NODES + iB];
    const int*   arowA = adj  + (size_t)iA * NODES;
    const int*   arowB = adj  + (size_t)iB * NODES;
    const float* drow  = adst + (size_t)h * NODES;

    v8f cA0 = {}, cA1 = {}, cA2 = {}, cA3 = {};
    v8f cB0 = {}, cB1 = {}, cB2 = {}, cB3 = {};

    for (int j0 = 0; j0 < NODES; j0 += 32) {
        // Shared per-column attention terms (same for both i-tiles).
        const float4* dp  = (const float4*)(drow + j0 + sel);
        const float4* dp2 = (const float4*)(drow + j0 + 16 + sel);
        float4 d0 = dp[0], d1 = dp[1], d2 = dp2[0], d3 = dp2[1];

        // Adjacency for both row tiles.
        const int4* apA  = (const int4*)(arowA + j0 + sel);
        const int4* apA2 = (const int4*)(arowA + j0 + 16 + sel);
        int4 aA0 = apA[0], aA1 = apA[1], aA2 = apA2[0], aA3 = apA2[1];
        const int4* apB  = (const int4*)(arowB + j0 + sel);
        const int4* apB2 = (const int4*)(arowB + j0 + 16 + sel);
        int4 aB0 = apB[0], aB1 = apB[1], aB2 = apB2[0], aB3 = apB2[1];

#define PV(AS_, M_, A_, D_) ((_Float16)((A_) ? __expf(lrelu((AS_) + (D_)) - (M_)) : 0.0f))
        v16h pA, pB;
        pA[0]  = PV(asA, mA, aA0.x, d0.x); pA[1]  = PV(asA, mA, aA0.y, d0.y);
        pA[2]  = PV(asA, mA, aA0.z, d0.z); pA[3]  = PV(asA, mA, aA0.w, d0.w);
        pA[4]  = PV(asA, mA, aA1.x, d1.x); pA[5]  = PV(asA, mA, aA1.y, d1.y);
        pA[6]  = PV(asA, mA, aA1.z, d1.z); pA[7]  = PV(asA, mA, aA1.w, d1.w);
        pA[8]  = PV(asA, mA, aA2.x, d2.x); pA[9]  = PV(asA, mA, aA2.y, d2.y);
        pA[10] = PV(asA, mA, aA2.z, d2.z); pA[11] = PV(asA, mA, aA2.w, d2.w);
        pA[12] = PV(asA, mA, aA3.x, d3.x); pA[13] = PV(asA, mA, aA3.y, d3.y);
        pA[14] = PV(asA, mA, aA3.z, d3.z); pA[15] = PV(asA, mA, aA3.w, d3.w);

        pB[0]  = PV(asB, mB, aB0.x, d0.x); pB[1]  = PV(asB, mB, aB0.y, d0.y);
        pB[2]  = PV(asB, mB, aB0.z, d0.z); pB[3]  = PV(asB, mB, aB0.w, d0.w);
        pB[4]  = PV(asB, mB, aB1.x, d1.x); pB[5]  = PV(asB, mB, aB1.y, d1.y);
        pB[6]  = PV(asB, mB, aB1.z, d1.z); pB[7]  = PV(asB, mB, aB1.w, d1.w);
        pB[8]  = PV(asB, mB, aB2.x, d2.x); pB[9]  = PV(asB, mB, aB2.y, d2.y);
        pB[10] = PV(asB, mB, aB2.z, d2.z); pB[11] = PV(asB, mB, aB2.w, d2.w);
        pB[12] = PV(asB, mB, aB3.x, d3.x); pB[13] = PV(asB, mB, aB3.y, d3.y);
        pB[14] = PV(asB, mB, aB3.z, d3.z); pB[15] = PV(asB, mB, aB3.w, d3.w);
#undef PV

        // B fragments: lane = column d, j = {sel..sel+7, 16+sel..16+sel+7} (two 16B loads).
        // Loaded once, used by both i-tiles (8 WMMAs per fragment set).
        const size_t joff = (size_t)j0 + sel;
        const _Float16* q0 = hT16 + (size_t)(h * OD +  0 + lr) * NODES + joff;
        const _Float16* q1 = hT16 + (size_t)(h * OD + 16 + lr) * NODES + joff;
        const _Float16* q2 = hT16 + (size_t)(h * OD + 32 + lr) * NODES + joff;
        const _Float16* q3 = hT16 + (size_t)(h * OD + 48 + lr) * NODES + joff;
        v16h b0 = __builtin_shufflevector(*(const v8h*)q0, *(const v8h*)(q0 + 16), 0,1,2,3,4,5,6,7,8,9,10,11,12,13,14,15);
        v16h b1 = __builtin_shufflevector(*(const v8h*)q1, *(const v8h*)(q1 + 16), 0,1,2,3,4,5,6,7,8,9,10,11,12,13,14,15);
        v16h b2 = __builtin_shufflevector(*(const v8h*)q2, *(const v8h*)(q2 + 16), 0,1,2,3,4,5,6,7,8,9,10,11,12,13,14,15);
        v16h b3 = __builtin_shufflevector(*(const v8h*)q3, *(const v8h*)(q3 + 16), 0,1,2,3,4,5,6,7,8,9,10,11,12,13,14,15);

        cA0 = __builtin_amdgcn_wmma_f32_16x16x32_f16(false, pA, false, b0, (short)0, cA0, false, false);
        cB0 = __builtin_amdgcn_wmma_f32_16x16x32_f16(false, pB, false, b0, (short)0, cB0, false, false);
        cA1 = __builtin_amdgcn_wmma_f32_16x16x32_f16(false, pA, false, b1, (short)0, cA1, false, false);
        cB1 = __builtin_amdgcn_wmma_f32_16x16x32_f16(false, pB, false, b1, (short)0, cB1, false, false);
        cA2 = __builtin_amdgcn_wmma_f32_16x16x32_f16(false, pA, false, b2, (short)0, cA2, false, false);
        cB2 = __builtin_amdgcn_wmma_f32_16x16x32_f16(false, pB, false, b2, (short)0, cB2, false, false);
        cA3 = __builtin_amdgcn_wmma_f32_16x16x32_f16(false, pA, false, b3, (short)0, cA3, false, false);
        cB3 = __builtin_amdgcn_wmma_f32_16x16x32_f16(false, pB, false, b3, (short)0, cB3, false, false);
    }

    // Row scales 1/s for the 8+8 rows this lane holds (consecutive runs).
    const float* rpA = rinv + (size_t)h * NODES + i0 + hi * 8;
    const float* rpB = rpA + 16;
    float4 rA0 = *(const float4*)rpA, rA1 = *(const float4*)(rpA + 4);
    float4 rB0 = *(const float4*)rpB, rB1 = *(const float4*)(rpB + 4);
    float rsA[8] = {rA0.x, rA0.y, rA0.z, rA0.w, rA1.x, rA1.y, rA1.z, rA1.w};
    float rsB[8] = {rB0.x, rB0.y, rB0.z, rB0.w, rB1.x, rB1.y, rB1.z, rB1.w};

    v8f accA[4] = {cA0, cA1, cA2, cA3};
    v8f accB[4] = {cB0, cB1, cB2, cB3};
    #pragma unroll
    for (int t = 0; t < 4; ++t) {
        #pragma unroll
        for (int v = 0; v < 8; ++v) {
            const int colo = h * OD + t * 16 + lr;
            out[(size_t)(i0 + hi * 8 + v) * HD + colo]      = accA[t][v] * rsA[v];
            out[(size_t)(i0 + 16 + hi * 8 + v) * HD + colo] = accB[t][v] * rsB[v];
        }
    }
}

// ---------------------------------------------------------------------------
extern "C" void kernel_launch(void* const* d_in, const int* in_sizes, int n_in,
                              void* d_out, int out_size, void* d_ws, size_t ws_size,
                              hipStream_t stream) {
    const float* x     = (const float*)d_in[0];
    const int*   adj   = (const int*)d_in[1];
    const float* W     = (const float*)d_in[2];
    const float* a_src = (const float*)d_in[3];
    const float* a_dst = (const float*)d_in[4];
    float* out = (float*)d_out;

    char* ws = (char*)d_ws;
    _Float16* hT16 = (_Float16*)ws;                                 // 2 MB
    float*    hT32 = (float*)(ws + (size_t)2 * 1024 * 1024);        // 4 MB
    float*    asrc = (float*)(ws + (size_t)6 * 1024 * 1024);        // 64 KB
    float*    adst = (float*)(ws + (size_t)6 * 1024 * 1024 + 64 * 1024);
    float*    mrow = (float*)(ws + (size_t)6 * 1024 * 1024 + 128 * 1024);
    float*    rin  = (float*)(ws + (size_t)6 * 1024 * 1024 + 192 * 1024);

    k_gemm_xw<<<dim3(1024), dim3(128), 0, stream>>>(x, W, hT16, hT32);
    k_alpha<<<dim3(NODES / 256), dim3(256), 0, stream>>>(hT32, a_src, a_dst, asrc, adst);
    k_stats<<<dim3(NODES), dim3(256), 0, stream>>>(adj, asrc, adst, mrow, rin);
    k_aggregate<<<dim3(NODES / 32), dim3(128), 0, stream>>>(adj, hT16, asrc, adst, mrow, rin, out);
}
